// Decoder_43138651521440
// MI455X (gfx1250) — matrix operand order repaired and checked by reading, not compiled, not run on hardware
//
#include <hip/hip_runtime.h>
#include <hip/hip_bf16.h>
#include <math.h>

// Problem dims (fixed by the reference)
#define BB   32      // batch
#define TT   128     // decode steps
#define SS   128     // encoder time
#define EE   512     // embed dim
#define HH   2048    // hidden
#define G4   8192    // 4*H gates
#define KC   2560    // fused K = E (x part) + H (h part)
#define CTXW 1024    // 2*enc_hidden

typedef __attribute__((ext_vector_type(8)))  float   v8f;
typedef __attribute__((ext_vector_type(16))) __bf16  v16bf;
typedef __attribute__((ext_vector_type(8)))  __bf16  v8bf;

// ---------------------------------------------------------------- prep kernels

// context[b][j] = max_s encoder_outs[s][b][j]
__global__ void ctx_max_kernel(const float* __restrict__ enc,
                               float* __restrict__ ctx) {
    int idx = blockIdx.x * blockDim.x + threadIdx.x;   // over B*1024
    if (idx >= BB * CTXW) return;
    float m = -INFINITY;
    for (int s = 0; s < SS; ++s)
        m = fmaxf(m, enc[(size_t)s * BB * CTXW + idx]);
    ctx[idx] = m;
}

// x_bf16[t][b][e] = bf16(embed[prev_tokens[b][t]][e])
__global__ void embed_gather_kernel(const float* __restrict__ embed,
                                    const int* __restrict__ prev,
                                    __bf16* __restrict__ xbf) {
    int idx = blockIdx.x * blockDim.x + threadIdx.x;   // over T*B*E
    if (idx >= TT * BB * EE) return;
    int e = idx & (EE - 1);
    int r = idx >> 9;          // = t*B + b
    int b = r & (BB - 1);
    int t = r >> 5;
    int tok = prev[b * TT + t];
    xbf[idx] = (__bf16)embed[(size_t)tok * EE + e];
}

// Wc[g][k] : k<512 -> W_ih[g][k] (x part), else W_hh[g][k-512]   (bf16)
__global__ void wconv_kernel(const float* __restrict__ W_ih,
                             const float* __restrict__ W_hh,
                             __bf16* __restrict__ Wc) {
    size_t idx = (size_t)blockIdx.x * blockDim.x + threadIdx.x; // over G4*KC
    if (idx >= (size_t)G4 * KC) return;
    int k = (int)(idx % KC);
    int g = (int)(idx / KC);
    float v = (k < EE) ? W_ih[(size_t)g * 2048 + k]
                       : W_hh[(size_t)g * HH + (k - EE)];
    Wc[idx] = (__bf16)v;
}

// static bias: sb[b][g] = b_ih[g] + b_hh[g]
//            + sum_k lang[b][k] * W_ih[g][512+k]      (k<512)
//            + sum_j ctx[b][j]  * W_ih[g][1024+j]     (j<1024)
__global__ void static_bias_kernel(const float* __restrict__ embed,
                                   const int* __restrict__ langs,
                                   const float* __restrict__ ctx,
                                   const float* __restrict__ W_ih,
                                   const float* __restrict__ b_ih,
                                   const float* __restrict__ b_hh,
                                   float* __restrict__ sb) {
    int idx = blockIdx.x * blockDim.x + threadIdx.x;   // over B*G4, b fastest
    if (idx >= BB * G4) return;
    int b = idx & (BB - 1);
    int g = idx >> 5;
    const float* wrow = W_ih + (size_t)g * 2048;
    const float* lang = embed + (size_t)langs[b] * EE;
    const float* cb   = ctx + b * CTXW;
    float acc = b_ih[g] + b_hh[g];
    for (int k = 0; k < EE; ++k)    acc += lang[k] * wrow[EE + k];
    for (int j = 0; j < CTXW; ++j)  acc += cb[j]   * wrow[1024 + j];
    sb[(size_t)b * G4 + g] = acc;
}

// h0 (bf16) and c0 (f32) from encoder state
__global__ void init_state_kernel(const float* __restrict__ ench,
                                  const float* __restrict__ encc,
                                  __bf16* __restrict__ h0,
                                  float* __restrict__ c0) {
    int idx = blockIdx.x * blockDim.x + threadIdx.x;   // over B*H
    if (idx >= BB * HH) return;
    h0[idx] = (__bf16)ench[idx];
    c0[idx] = encc[idx];
}

// ---------------------------------------------------------------- LSTM step

__device__ __forceinline__ v16bf cat16(v8bf lo, v8bf hi) {
    return __builtin_shufflevector(lo, hi,
        0, 1, 2, 3, 4, 5, 6, 7, 8, 9, 10, 11, 12, 13, 14, 15);
}
__device__ __forceinline__ float sigm(float x) {
    return 1.0f / (1.0f + __expf(-x));
}

// One wave = one 16-wide h-tile; computes {2 M-tiles} x {i,f,g,o} WMMA tiles,
// then does the LSTM pointwise update fully in registers.
__global__ __launch_bounds__(32)
void lstm_step_kernel(const __bf16* __restrict__ Wc,   // [G4][KC]
                      const __bf16* __restrict__ xt,   // [B][E]  (this step)
                      const __bf16* __restrict__ h_in, // [B][H]
                      const float*  __restrict__ sb,   // [B][G4]
                      float*        __restrict__ cbuf, // [B][H]
                      float*        __restrict__ out_t,// [B][H]  (d_out slice)
                      __bf16*       __restrict__ h_out)// [B][H]
{
    const int ht   = blockIdx.x;       // 0..127 h-tile
    const int lane = threadIdx.x;      // 0..31
    const int ln   = lane & 15;
    const int hh   = lane >> 4;        // lane half (ISA A/B K-split)

    v8f z = {};
    v8f acc[2][4];
    #pragma unroll
    for (int mt = 0; mt < 2; ++mt)
        #pragma unroll
        for (int gi = 0; gi < 4; ++gi) acc[mt][gi] = z;

    // B-matrix: lane ln = column N; lanes 0-15 take K[k0..+7] and K[k0+16..+23],
    // 16-31 take K[k0+8..+15] and K[k0+24..+31]; two 16B loads per tile.
    const __bf16* brow[4];
    #pragma unroll
    for (int gi = 0; gi < 4; ++gi) {
        int g = gi * 2048 + ht * 16 + ln;
        brow[gi] = Wc + (size_t)g * KC + hh * 8;
    }

    // A-matrix: lanes 0-15 hold rows M=0..15 with K[k0..+7] and K[k0+16..+23];
    // lanes 16-31 hold K[k0+8..+15] and K[k0+24..+31].
    const int aoff = hh * 8;

    // ---- phase 1: x contribution (K = 0..511) ----
    {
        const __bf16* a0 = xt + (size_t)(ln)      * EE;
        const __bf16* a1 = xt + (size_t)(16 + ln) * EE;
        for (int kk = 0; kk < 16; ++kk) {
            int k0 = kk * 32;
            v16bf A0 = cat16(*(const v8bf*)(a0 + k0 + aoff),
                             *(const v8bf*)(a0 + k0 + aoff + 16));
            v16bf A1 = cat16(*(const v8bf*)(a1 + k0 + aoff),
                             *(const v8bf*)(a1 + k0 + aoff + 16));
            #pragma unroll
            for (int gi = 0; gi < 4; ++gi) {
                v16bf Bm = cat16(*(const v8bf*)(brow[gi] + k0),
                                 *(const v8bf*)(brow[gi] + k0 + 16));
                acc[0][gi] = __builtin_amdgcn_wmma_f32_16x16x32_bf16(
                    false, A0, false, Bm, (short)0, acc[0][gi], false, false);
                acc[1][gi] = __builtin_amdgcn_wmma_f32_16x16x32_bf16(
                    false, A1, false, Bm, (short)0, acc[1][gi], false, false);
            }
        }
    }

    // ---- phase 2: recurrent h contribution (K = 512..2559) ----
    {
        const __bf16* a0 = h_in + (size_t)(ln)      * HH;
        const __bf16* a1 = h_in + (size_t)(16 + ln) * HH;
        for (int kk = 16; kk < 80; ++kk) {
            int k0 = kk * 32;
            int kh = k0 - EE;
            v16bf A0 = cat16(*(const v8bf*)(a0 + kh + aoff),
                             *(const v8bf*)(a0 + kh + aoff + 16));
            v16bf A1 = cat16(*(const v8bf*)(a1 + kh + aoff),
                             *(const v8bf*)(a1 + kh + aoff + 16));
            #pragma unroll
            for (int gi = 0; gi < 4; ++gi) {
                v16bf Bm = cat16(*(const v8bf*)(brow[gi] + k0),
                                 *(const v8bf*)(brow[gi] + k0 + 16));
                acc[0][gi] = __builtin_amdgcn_wmma_f32_16x16x32_bf16(
                    false, A0, false, Bm, (short)0, acc[0][gi], false, false);
                acc[1][gi] = __builtin_amdgcn_wmma_f32_16x16x32_bf16(
                    false, A1, false, Bm, (short)0, acc[1][gi], false, false);
            }
        }
    }

    // ---- pointwise LSTM update ----
    // C/D layout: VGPR r -> M = r (lanes 0-15) / r+8 (lanes 16-31), N = ln.
    const int h = ht * 16 + ln;
    #pragma unroll
    for (int mt = 0; mt < 2; ++mt) {
        #pragma unroll
        for (int r = 0; r < 8; ++r) {
            int b = mt * 16 + r + hh * 8;
            const float* sbb = sb + (size_t)b * G4 + h;
            float gi_ = acc[mt][0][r] + sbb[0];
            float gf  = acc[mt][1][r] + sbb[2048];
            float gg  = acc[mt][2][r] + sbb[4096];
            float go  = acc[mt][3][r] + sbb[6144];
            size_t bh = (size_t)b * HH + h;
            float cv  = sigm(gf) * cbuf[bh] + sigm(gi_) * tanhf(gg);
            float hv  = sigm(go) * tanhf(cv);
            cbuf[bh]  = cv;
            out_t[bh] = hv;
            h_out[bh] = (__bf16)hv;
        }
    }
}

// ---------------------------------------------------------------- launch

extern "C" void kernel_launch(void* const* d_in, const int* in_sizes, int n_in,
                              void* d_out, int out_size, void* d_ws, size_t ws_size,
                              hipStream_t stream) {
    const float* embed  = (const float*)d_in[0];
    const float* encout = (const float*)d_in[1];
    const float* ench   = (const float*)d_in[2];
    const float* encc   = (const float*)d_in[3];
    const float* W_ih   = (const float*)d_in[4];
    const float* W_hh   = (const float*)d_in[5];
    const float* b_ih   = (const float*)d_in[6];
    const float* b_hh   = (const float*)d_in[7];
    const int*   prev   = (const int*)d_in[8];
    const int*   langs  = (const int*)d_in[9];
    float* out = (float*)d_out;

    // workspace carve-up (~48 MB total)
    char* ws = (char*)d_ws;
    float*  ctx  = (float*)ws;   ws += (size_t)BB * CTXW * 4;
    float*  sb   = (float*)ws;   ws += (size_t)BB * G4 * 4;
    float*  cbuf = (float*)ws;   ws += (size_t)BB * HH * 4;
    __bf16* xbf  = (__bf16*)ws;  ws += (size_t)TT * BB * EE * 2;
    __bf16* hA   = (__bf16*)ws;  ws += (size_t)BB * HH * 2;
    __bf16* hB   = (__bf16*)ws;  ws += (size_t)BB * HH * 2;
    __bf16* Wc   = (__bf16*)ws;  ws += (size_t)G4 * KC * 2;

    ctx_max_kernel<<<(BB * CTXW) / 256, 256, 0, stream>>>(encout, ctx);
    embed_gather_kernel<<<(TT * BB * EE) / 256, 256, 0, stream>>>(embed, prev, xbf);
    wconv_kernel<<<(int)(((size_t)G4 * KC) / 256), 256, 0, stream>>>(W_ih, W_hh, Wc);
    static_bias_kernel<<<(BB * G4) / 256, 256, 0, stream>>>(embed, langs, ctx,
                                                            W_ih, b_ih, b_hh, sb);
    init_state_kernel<<<(BB * HH) / 256, 256, 0, stream>>>(ench, encc, hA, cbuf);

    // sequential scan: stream ordering serializes the 128 steps
    for (int t = 0; t < TT; ++t) {
        const __bf16* hin  = (t & 1) ? hB : hA;
        __bf16*       hout = (t & 1) ? hA : hB;
        lstm_step_kernel<<<HH / 16, 32, 0, stream>>>(
            Wc, xbf + (size_t)t * BB * EE, hin, sb, cbuf,
            out + (size_t)t * BB * HH, hout);
    }
}
